// Attn_15822659519022
// MI455X (gfx1250) — matrix operand order repaired and checked by reading, not compiled, not run on hardware
//
#include <hip/hip_runtime.h>
#include <cstdint>

typedef __attribute__((ext_vector_type(16))) _Float16 v16h;
typedef __attribute__((ext_vector_type(8)))  float    v8f;
typedef __attribute__((ext_vector_type(4)))  int      v4i;

#if defined(__HIP_DEVICE_COMPILE__) && __has_builtin(__builtin_amdgcn_global_load_async_to_lds_b128)
#define HAS_ASYNC_LDS 1
#else
#define HAS_ASYNC_LDS 0
#endif

#if HAS_ASYNC_LDS
typedef __attribute__((address_space(1))) v4i v4i_gbl;   // global-AS int4
typedef __attribute__((address_space(3))) v4i v4i_lds;   // LDS-AS int4
#endif

static __device__ __forceinline__ uint32_t pack2h(float a, float b) {
  union { _Float16 h[2]; uint32_t u; } p;
  p.h[0] = (_Float16)a; p.h[1] = (_Float16)b;
  return p.u;
}

// Copy 8 f16 (16 B) global -> LDS. Async (ASYNCcnt) when available, else direct.
static __device__ __forceinline__ void stage16(_Float16* dst, const _Float16* src) {
#if HAS_ASYNC_LDS
  __builtin_amdgcn_global_load_async_to_lds_b128((v4i_gbl*)src, (v4i_lds*)dst, 0, 0);
#else
  *(uint4*)dst = *(const uint4*)src;
#endif
}
static __device__ __forceinline__ void async_wait0() {
#if HAS_ASYNC_LDS
  asm volatile("s_wait_asynccnt 0x0" ::: "memory");
#endif
}

// ---- DPP16 cross-lane reductions over each 16-lane group (VALU, no LDS traffic) ----
template <int CTRL>
static __device__ __forceinline__ float dppmov(float x) {
  return __builtin_bit_cast(float,
      __builtin_amdgcn_update_dpp(0, __builtin_bit_cast(int, x), CTRL, 0xf, 0xf, true));
}
static __device__ __forceinline__ float row16_max(float x) {
  x = fmaxf(x, dppmov<0xB1>(x));    // quad_perm [1,0,3,2]  (xor1)
  x = fmaxf(x, dppmov<0x4E>(x));    // quad_perm [2,3,0,1]  (xor2)
  x = fmaxf(x, dppmov<0x141>(x));   // ROW_HALF_MIRROR      (xor7: join quads)
  x = fmaxf(x, dppmov<0x128>(x));   // ROW_ROR:8            (join 8-groups)
  return x;
}
static __device__ __forceinline__ float row16_sum(float x) {
  x += dppmov<0xB1>(x);
  x += dppmov<0x4E>(x);
  x += dppmov<0x141>(x);
  x += dppmov<0x128>(x);
  return x;
}

// A fragment (16x32 f16, MxK, ISA 7.12.2): lane=row; halves [8hi,8hi+8) and [16+8hi,+8).
static __device__ __forceinline__ v16h frag_a16(const _Float16* p, int hi) {
  union { v16h v; uint4 q[2]; } f;
  f.q[0] = *(const uint4*)(p + 8 * hi);
  f.q[1] = *(const uint4*)(p + 16 + 8 * hi);
  return f.v;
}
// B fragment (32x16 f16, KxN): lane=col; halves [8hi,8hi+8) and [16+8hi,+8). p at column start (K-major).
static __device__ __forceinline__ v16h frag_b16(const _Float16* p, int hi) {
  union { v16h v; uint4 q[2]; } f;
  f.q[0] = *(const uint4*)(p + 8 * hi);
  f.q[1] = *(const uint4*)(p + 16 + 8 * hi);
  return f.v;
}
static __device__ __forceinline__ v8f zero8() {
  v8f z;
#pragma unroll
  for (int i = 0; i < 8; ++i) z[i] = 0.0f;
  return z;
}
#define WMMA_F16(A_, B_, C_) \
  __builtin_amdgcn_wmma_f32_16x16x32_f16(false, (A_), false, (B_), (short)0, (C_), false, false)

// ---------------- one-time converts ----------------
__global__ __launch_bounds__(256) void cvt_f32_f16(const float* __restrict__ in,
                                                   _Float16* __restrict__ out, int n4) {
  int i = blockIdx.x * blockDim.x + threadIdx.x;
  if (i < n4) {
    float4 f = ((const float4*)in)[i];
    union { _Float16 h[4]; uint2 u; } p;
    p.h[0] = (_Float16)f.x; p.h[1] = (_Float16)f.y;
    p.h[2] = (_Float16)f.z; p.h[3] = (_Float16)f.w;
    ((uint2*)out)[i] = p.u;
  }
}
// wt[n*K + k] = w[k*N + n] * (n < scaleN ? 0.125 : 1)   (fold 1/sqrt(HD) into Q columns)
__global__ __launch_bounds__(256) void transpose_cvt(const float* __restrict__ w,
                                                     _Float16* __restrict__ wt,
                                                     int K, int N, int scaleN) {
  int idx = blockIdx.x * blockDim.x + threadIdx.x;
  if (idx < N * K) {
    int n = idx / K, k = idx - n * K;
    float v = w[(size_t)k * N + n];
    if (n < scaleN) v *= 0.125f;
    wt[idx] = (_Float16)v;
  }
}

// ---------------- f16 GEMM: C = A[M,K] * B, with Bt = B^T row-major [N][K] ----------------
constexpr int MODE_QKV = 1;   // C0 = qkv f16 [tok][3072] (q|k part), C1 = v transposed [b*D+hd][T]
constexpr int MODE_F32 = 2;   // C0 = f32 row-major [M][N]

template <int MODE>
__global__ __launch_bounds__(256) void gemm16(
    const _Float16* __restrict__ A, const _Float16* __restrict__ Bt,
    void* __restrict__ C0, void* __restrict__ C1, int M, int N, int K)
{
  (void)M; (void)C1;
  constexpr int BM = 128, BN = 128, BK = 64, P = 72;   // P*2B = 144 B pitch, 16B-aligned
  __shared__ _Float16 As[2][BM * P];                   // [row][k]
  __shared__ _Float16 Bs[2][BN * P];                   // [n][k]
  const int tid = threadIdx.x;
  const int lane = tid & 31, wave = tid >> 5;
  const int ln = lane & 15, hi = lane >> 4;
  const int wm = wave & 1, wn = wave >> 1;             // 2 x 4 wave grid
  const int m0 = blockIdx.y * BM, n0 = blockIdx.x * BN;

  v8f acc[4][2];
#pragma unroll
  for (int mi = 0; mi < 4; ++mi)
#pragma unroll
    for (int ni = 0; ni < 2; ++ni) acc[mi][ni] = zero8();

  const int NT = K / BK;
  auto stageTile = [&](int buf, int kt) {
#pragma unroll
    for (int i = 0; i < 4; ++i) {
      int lin = tid + i * 256;
      int row = lin >> 3;            // 8 x 16B per 64-half row
      int hc  = (lin & 7) << 3;
      stage16(&As[buf][row * P + hc], A  + (size_t)(m0 + row) * K + kt + hc);
      stage16(&Bs[buf][row * P + hc], Bt + (size_t)(n0 + row) * K + kt + hc);
    }
  };

  stageTile(0, 0);
  for (int kt = 0; kt < NT; ++kt) {
    const int cur = kt & 1;
    async_wait0();
    __syncthreads();
    if (kt + 1 < NT) stageTile(1 - cur, (kt + 1) * BK);
#pragma unroll
    for (int kk = 0; kk < 2; ++kk) {
      v16h af[4], bf[2];
#pragma unroll
      for (int mi = 0; mi < 4; ++mi)
        af[mi] = frag_a16(&As[cur][(wm * 64 + mi * 16 + ln) * P + kk * 32], hi);
#pragma unroll
      for (int ni = 0; ni < 2; ++ni)
        bf[ni] = frag_b16(&Bs[cur][(wn * 32 + ni * 16 + ln) * P + kk * 32], hi);
#pragma unroll
      for (int mi = 0; mi < 4; ++mi)
#pragma unroll
        for (int ni = 0; ni < 2; ++ni)
          acc[mi][ni] = WMMA_F16(af[mi], bf[ni], acc[mi][ni]);
    }
  }

  // C/D layout: vgpr r -> M = r + 8*hi; N = lane&15
  if constexpr (MODE == MODE_QKV) {
    _Float16* qk = (_Float16*)C0;
    _Float16* vt = (_Float16*)C1;
    if (n0 < 2048) {                                   // q|k columns: row-major f16
#pragma unroll
      for (int mi = 0; mi < 4; ++mi) {
        int row0 = m0 + wm * 64 + mi * 16 + 8 * hi;
#pragma unroll
        for (int r = 0; r < 8; ++r) {
          _Float16* p = qk + (size_t)(row0 + r) * 3072 + n0 + wn * 32 + ln;
          p[0]  = (_Float16)acc[mi][0][r];
          p[16] = (_Float16)acc[mi][1][r];
        }
      }
    } else {                                           // v columns: transposed [b*D + hd][T]
      int colb = n0 - 2048 + wn * 32 + ln;
#pragma unroll
      for (int mi = 0; mi < 4; ++mi) {
        int row0 = m0 + wm * 64 + mi * 16 + 8 * hi;
        int bb = row0 >> 11, t0 = row0 & 2047;
#pragma unroll
        for (int ni = 0; ni < 2; ++ni) {
          _Float16* p = vt + (size_t)(bb * 1024 + colb + ni * 16) * 2048 + t0;
#pragma unroll
          for (int r = 0; r < 8; r += 2)
            *(uint32_t*)(p + r) = pack2h(acc[mi][ni][r], acc[mi][ni][r + 1]);
        }
      }
    }
  } else {
    float* Cf = (float*)C0;
#pragma unroll
    for (int mi = 0; mi < 4; ++mi) {
      int row0 = m0 + wm * 64 + mi * 16 + 8 * hi;
#pragma unroll
      for (int r = 0; r < 8; ++r) {
        float* p = Cf + (size_t)(row0 + r) * N + n0 + wn * 32 + ln;
        p[0]  = acc[mi][0][r];
        p[16] = acc[mi][1][r];
      }
    }
  }
}

// ---------------- causal flash attention (f16 in, f16 out) ----------------
// qkvh: [tok][3072] f16 (q pre-scaled via W; k at +1024). vt: [b*1024 + h*64 + d][2048] f16.
__global__ __launch_bounds__(128) void attn_fa(const _Float16* __restrict__ qkvh,
                                               const _Float16* __restrict__ vt,
                                               _Float16* __restrict__ outh)
{
  constexpr int T = 2048, D = 1024, S3 = 3072, P = 72;
  __shared__ _Float16 Ks[2][64 * P];   // [key][d]
  __shared__ _Float16 Vs[2][64 * P];   // [d][key]
  __shared__ _Float16 Ps[4][16 * P];   // per-wave P tile [row][key]

  const int qb = blockIdx.x, h = blockIdx.y, b = blockIdx.z;
  const int wave = threadIdx.x >> 5, lane = threadIdx.x & 31;
  const int ln = lane & 15, hi = lane >> 4;
  const int q0 = qb * 64 + wave * 16;

  // Q fragments straight from f16 global (scale already folded into Wq)
  v16h aQ[2];
  {
    const _Float16* qp = qkvh + (size_t)(b * T + q0 + ln) * S3 + h * 64;
#pragma unroll
    for (int kk = 0; kk < 2; ++kk) {
      union { v16h v; uint4 q[2]; } f;
      f.q[0] = *(const uint4*)(qp + kk * 32 + 8 * hi);
      f.q[1] = *(const uint4*)(qp + kk * 32 + 16 + 8 * hi);
      aQ[kk] = f.v;
    }
  }

  float mrun[8], lrun[8];
  v8f Oac[4];
#pragma unroll
  for (int r = 0; r < 8; ++r) { mrun[r] = -1e30f; lrun[r] = 0.0f; }
#pragma unroll
  for (int nd = 0; nd < 4; ++nd) Oac[nd] = zero8();

  auto stageKV = [&](int buf, int kb) {
#pragma unroll
    for (int i = 0; i < 4; ++i) {
      int lin = threadIdx.x + i * 128;
      int row = lin >> 3;            // 64 rows x 8 x 16B
      int hc  = (lin & 7) << 3;
      stage16(&Ks[buf][row * P + hc],
              qkvh + (size_t)(b * T + kb * 64 + row) * S3 + D + h * 64 + hc);
      stage16(&Vs[buf][row * P + hc],
              vt + (size_t)(b * 1024 + h * 64 + row) * T + kb * 64 + hc);
    }
  };

  stageKV(0, 0);
  for (int kb = 0; kb <= qb; ++kb) {
    const int cur = kb & 1;
    async_wait0();
    __syncthreads();
    if (kb + 1 <= qb) stageKV(1 - cur, kb + 1);

    // S[16q x 64k] = Q * K^T
    v8f Sac[4];
#pragma unroll
    for (int ni = 0; ni < 4; ++ni) Sac[ni] = zero8();
#pragma unroll
    for (int kk = 0; kk < 2; ++kk)
#pragma unroll
      for (int ni = 0; ni < 4; ++ni)
        Sac[ni] = WMMA_F16(aQ[kk], frag_b16(&Ks[cur][(ni * 16 + ln) * P + kk * 32], hi), Sac[ni]);

    if (kb == qb) {                  // causal mask only on the diagonal block
#pragma unroll
      for (int ni = 0; ni < 4; ++ni)
#pragma unroll
        for (int r = 0; r < 8; ++r) {
          int key = kb * 64 + ni * 16 + ln;
          int qr  = q0 + r + 8 * hi;
          if (key > qr) Sac[ni][r] = -1e9f;
        }
    }

    // online softmax: rows in vgpr r (+8 per lane half), cols in 16-lane groups (DPP reduce)
#pragma unroll
    for (int r = 0; r < 8; ++r) {
      float m4 = fmaxf(fmaxf(Sac[0][r], Sac[1][r]), fmaxf(Sac[2][r], Sac[3][r]));
      m4 = row16_max(m4);
      float mn = fmaxf(mrun[r], m4);
      float sc = __expf(mrun[r] - mn);
      mrun[r] = mn;
      lrun[r] *= sc;
#pragma unroll
      for (int nd = 0; nd < 4; ++nd) Oac[nd][r] *= sc;
      float ps = 0.0f;
#pragma unroll
      for (int ni = 0; ni < 4; ++ni) {
        float p = __expf(Sac[ni][r] - mn);
        Sac[ni][r] = p;
        ps += p;
      }
      lrun[r] += row16_sum(ps);
    }

    // P -> per-wave LDS tile (C-frag -> A-frag re-layout); same wave, no barrier needed
#pragma unroll
    for (int ni = 0; ni < 4; ++ni)
#pragma unroll
      for (int r = 0; r < 8; ++r)
        Ps[wave][(r + 8 * hi) * P + ni * 16 + ln] = (_Float16)Sac[ni][r];

    // O += P * V
#pragma unroll
    for (int kk = 0; kk < 2; ++kk) {
      v16h aP = frag_a16(&Ps[wave][ln * P + kk * 32], hi);
#pragma unroll
      for (int nd = 0; nd < 4; ++nd)
        Oac[nd] = WMMA_F16(aP, frag_b16(&Vs[cur][(nd * 16 + ln) * P + kk * 32], hi), Oac[nd]);
    }
  }

#pragma unroll
  for (int r = 0; r < 8; ++r) {
    float inv = 1.0f / lrun[r];
    int row = q0 + r + 8 * hi;
    _Float16* p = outh + (size_t)(b * T + row) * D + h * 64 + ln;
#pragma unroll
    for (int nd = 0; nd < 4; ++nd) p[nd * 16] = (_Float16)(Oac[nd][r] * inv);
  }
}

extern "C" void kernel_launch(void* const* d_in, const int* in_sizes, int n_in,
                              void* d_out, int out_size, void* d_ws, size_t ws_size,
                              hipStream_t stream) {
  (void)in_sizes; (void)n_in; (void)out_size; (void)ws_size;
  const float* x  = (const float*)d_in[0];   // [2,2048,1024]
  const float* wq = (const float*)d_in[1];   // [1024,3072]
  const float* wp = (const float*)d_in[2];   // [1024,1024]
  float* out = (float*)d_out;                // [2,2048,1024] f32

  _Float16* xh     = (_Float16*)d_ws;                    // 4096*1024
  _Float16* wqkvT  = xh + (size_t)4096 * 1024;           // 3072*1024 (Q cols pre-scaled)
  _Float16* wprojT = wqkvT + (size_t)3072 * 1024;        // 1024*1024
  _Float16* qkvh   = wprojT + (size_t)1024 * 1024;       // 4096*3072 (q|k used)
  _Float16* vth    = qkvh + (size_t)4096 * 3072;         // 2048*2048 ([b*D+hd][T])
  _Float16* atth   = vth + (size_t)2048 * 2048;          // 4096*1024

  cvt_f32_f16<<<4096, 256, 0, stream>>>(x, xh, 4096 * 1024 / 4);
  transpose_cvt<<<(3072 * 1024 + 255) / 256, 256, 0, stream>>>(wq, wqkvT, 1024, 3072, 1024);
  transpose_cvt<<<(1024 * 1024 + 255) / 256, 256, 0, stream>>>(wp, wprojT, 1024, 1024, 0);
  {
    dim3 g(3072 / 128, 4096 / 128);
    gemm16<MODE_QKV><<<g, 256, 0, stream>>>(xh, wqkvT, qkvh, vth, 4096, 3072, 1024);
  }
  {
    dim3 g(2048 / 64, 16, 2);
    attn_fa<<<g, 128, 0, stream>>>(qkvh, vth, atth);
  }
  {
    dim3 g(1024 / 128, 4096 / 128);
    gemm16<MODE_F32><<<g, 256, 0, stream>>>(atth, wprojT, out, nullptr, 4096, 1024, 1024);
  }
}
